// PGC_31396210933771
// MI455X (gfx1250) — matrix operand order repaired and checked by reading, not compiled, not run on hardware
//
#include <hip/hip_runtime.h>

// ============================================================================
// Gated conv block for MI455X (gfx1250, wave32, WMMA).
//
// Roofline: ~103 GFLOP, ~270 MB min HBM traffic -> ~12us memory-bound floor
// at 23.3 TB/s; bf16 WMMA compute lands in the same range. Strategy:
// v_wmma_f32_16x16x32_bf16 with f32 accumulation, full fusion (bias, RMSNorm,
// depthwise conv, gating inside the two GEMM kernels), one bf16 intermediate.
//
// Round-3 observation: the post-RA scheduler serializes each B load-pair with
// its WMMA (s_wait_loadcnt 0x0) no matter how the source pipelines registers.
// The effective latency-hiding mechanism is wave interleaving, so round 4
// halves per-wave VGPRs (8 frags -> 64 accum VGPRs in K1, 4 frags in K2) and
// doubles waves per workgroup (1024 threads): 8 waves/SIMD round-robin over
// the load->WMMA dependency chain instead of 4.
// ============================================================================

#define D_MODEL   1024
#define EXPANDED  1024
#define TWO_E     2048
#define SEQ_T     4096
#define BT_TOTAL  16384     // b*t = 4*4096
#define EPSILON   1e-8f

typedef __attribute__((ext_vector_type(16))) __bf16 v16bf;
typedef __attribute__((ext_vector_type(8)))  __bf16 v8bf;
typedef __attribute__((ext_vector_type(8)))  float  v8f;

// ---------------------------------------------------------------------------
// B-matrix fragment (32x16, 16-bit): lane n = l&15 holds 16 contiguous K
// values of column n; lanes 0-15 -> K[0..15], lanes 16-31 -> K[16..31].
// Since C = A * W^T, column n of B == row n of W: contiguous row-major load.
// ---------------------------------------------------------------------------
__device__ __forceinline__ v16bf load_b_frag(const __bf16* __restrict__ p) {
  v8bf lo = *(const v8bf*)(p);
  v8bf hi = *(const v8bf*)(p + 8);
  v16bf r;
#pragma unroll
  for (int i = 0; i < 8; ++i) { r[i] = lo[i]; r[8 + i] = hi[i]; }
  return r;
}

// ---------------------------------------------------------------------------
// A-matrix fragment (16x32, 16-bit) from an f32 row with on-the-fly bf16 cvt.
// Lane l<16 holds row M=l, elems 0..15 = K[0..15];
// lane l>=16 holds same row, K[16..31]  (khalf = lane>>4).
// ---------------------------------------------------------------------------
__device__ __forceinline__ v16bf make_a_frag_f32(const float* __restrict__ rowp,
                                                 int khalf) {
  const float4* q0 = (const float4*)(rowp + khalf * 16);
  const float4* q1 = (const float4*)(rowp + khalf * 16 + 8);
  float4 a0 = q0[0], a1 = q0[1], b0 = q1[0], b1 = q1[1];
  v16bf r;
  r[0]  = (__bf16)a0.x; r[1]  = (__bf16)a0.y; r[2]  = (__bf16)a0.z; r[3]  = (__bf16)a0.w;
  r[4]  = (__bf16)a1.x; r[5]  = (__bf16)a1.y; r[6]  = (__bf16)a1.z; r[7]  = (__bf16)a1.w;
  r[8]  = (__bf16)b0.x; r[9]  = (__bf16)b0.y; r[10] = (__bf16)b0.z; r[11] = (__bf16)b0.w;
  r[12] = (__bf16)b1.x; r[13] = (__bf16)b1.y; r[14] = (__bf16)b1.z; r[15] = (__bf16)b1.w;
  return r;
}

// ============================================================================
// K0: one-shot weight conversion f32 -> bf16 into workspace (weights are
// re-read by every workgroup; bf16 halves their L2 footprint, 6 MB total,
// fully resident in the 192 MB L2).
// ============================================================================
__global__ void k_cvt_weights(const float* __restrict__ Win,
                              const float* __restrict__ Wout,
                              __bf16* __restrict__ WinB,
                              __bf16* __restrict__ WoutB) {
  int i = blockIdx.x * blockDim.x + threadIdx.x;
  const int nWin = TWO_E * D_MODEL;
  if (i < nWin) {
    WinB[i] = (__bf16)Win[i];
  } else {
    int j = i - nWin;
    if (j < D_MODEL * EXPANDED) WoutB[j] = (__bf16)Wout[j];
  }
}

// ============================================================================
// K1: xv = rmsnorm(u @ W_in^T + b_in) * in_norm_w   (written as bf16)
// Workgroup: 32 rows x full 2048 cols (RMSNorm reduces the whole row).
// 1024 threads = 32 waves, wave grid 2(M) x 16(N); wave tile 16 x 128
// -> 8 fragments = 64 accum VGPRs/lane (~110 total, under the 128 cap),
// giving 8 waves/SIMD to interleave over the load->WMMA chain.
// ============================================================================
__global__ void __launch_bounds__(1024)
k_inproj(const float*  __restrict__ u,
         const __bf16* __restrict__ WinB,
         const float*  __restrict__ b_in,
         const float*  __restrict__ nrm_w,
         __bf16*       __restrict__ xv) {
  __shared__ float s_ss[32];

  const int tid   = threadIdx.x;
  const int lane  = tid & 31;
  const int wave  = tid >> 5;       // 0..31
  const int mw    = wave >> 4;      // 0..1   (16-row group)
  const int nw    = wave & 15;      // 0..15  (128-col group)
  const int l15   = lane & 15;
  const int khalf = lane >> 4;
  const int m0    = blockIdx.x * 32;
  const int nb    = nw * 128;

  v8f acc[8];
#pragma unroll
  for (int f = 0; f < 8; ++f)
#pragma unroll
    for (int r = 0; r < 8; ++r) acc[f][r] = 0.f;

  const float*  uprow = u + (size_t)(m0 + mw * 16 + l15) * D_MODEL;
  const __bf16* wbase = WinB + (size_t)(nb + l15) * D_MODEL + khalf * 16;
  const size_t  fstride = (size_t)16 * D_MODEL;   // 16 W rows per fragment

#pragma unroll 1
  for (int k0 = 0; k0 < D_MODEL; k0 += 32) {
    __builtin_prefetch(uprow + k0 + 32, 0, 1);        // global_prefetch_b8
    v16bf a = make_a_frag_f32(uprow + k0, khalf);
    const __bf16* wb = wbase + k0;
    // B fragments pipelined in groups of 2
    v16bf bc0 = load_b_frag(wb);
    v16bf bc1 = load_b_frag(wb + fstride);
#pragma unroll
    for (int g = 0; g < 4; ++g) {
      v16bf bn0, bn1;
      if (g < 3) {
        bn0 = load_b_frag(wb + (size_t)(2 * g + 2) * fstride);
        bn1 = load_b_frag(wb + (size_t)(2 * g + 3) * fstride);
      }
      acc[2 * g] = __builtin_amdgcn_wmma_f32_16x16x32_bf16(
          false, a, false, bc0, (short)0, acc[2 * g], false, false);
      acc[2 * g + 1] = __builtin_amdgcn_wmma_f32_16x16x32_bf16(
          false, a, false, bc1, (short)0, acc[2 * g + 1], false, false);
      if (g < 3) { bc0 = bn0; bc1 = bn1; }
    }
  }

  // ---- epilogue: bias, row sum-of-squares, cross-wave reduce, scale, store
  if (tid < 32) s_ss[tid] = 0.f;
  __syncthreads();

  float ss[8];
#pragma unroll
  for (int r = 0; r < 8; ++r) ss[r] = 0.f;
#pragma unroll
  for (int f = 0; f < 8; ++f) {
    const float bias = b_in[nb + f * 16 + l15];
#pragma unroll
    for (int r = 0; r < 8; ++r) {
      float vv = acc[f][r] + bias;
      acc[f][r] = vv;
      ss[r] += vv * vv;
    }
  }
  // reduce across the 16 columns held in this lane's 16-lane half
#pragma unroll
  for (int off = 1; off < 16; off <<= 1)
#pragma unroll
    for (int r = 0; r < 8; ++r) ss[r] += __shfl_xor(ss[r], off, 32);

  const int rbase = mw * 16 + khalf * 8;   // first C/D row of this lane-half
  if (l15 == 0) {
#pragma unroll
    for (int r = 0; r < 8; ++r) atomicAdd(&s_ss[rbase + r], ss[r]);
  }
  __syncthreads();
  if (tid < 32) s_ss[tid] = rsqrtf(s_ss[tid] * (1.f / TWO_E) + EPSILON);
  __syncthreads();

  float scale[8];
#pragma unroll
  for (int r = 0; r < 8; ++r) scale[r] = s_ss[rbase + r];

#pragma unroll
  for (int f = 0; f < 8; ++f) {
    const int col = nb + f * 16 + l15;
    const float g = nrm_w[col];
#pragma unroll
    for (int r = 0; r < 8; ++r) {
      xv[(size_t)(m0 + rbase + r) * TWO_E + col] =
          (__bf16)(acc[f][r] * scale[r] * g);
    }
  }
}

// ============================================================================
// K2: out = rmsnorm( (v * depthwise_conv3(x)) @ W_out^T + b_out ) * out_norm_w
// The K=3 depthwise conv mixes only time, so it fuses into the A-operand LDS
// staging: each 32x32 gated chunk is computed ONCE (zero-padded at batch
// edges), stored as bf16 in LDS, and consumed via ds_load A-fragments.
// 1024 threads = 32 waves, wave grid 2(M) x 16(N); wave tile 16 x 64
// -> 4 fragments = 32 accum VGPRs/lane; all 4 B frags loaded before WMMAs.
// ============================================================================
__global__ void __launch_bounds__(1024)
k_outproj(const __bf16* __restrict__ xv,
          const __bf16* __restrict__ WoutB,
          const float*  __restrict__ conv_w,
          const float*  __restrict__ conv_b,
          const float*  __restrict__ b_out,
          const float*  __restrict__ nrm_w,
          float*        __restrict__ out) {
  __shared__ __bf16 s_g[32 * 32];   // gated A chunk, 2 KB
  __shared__ float  s_ss[32];

  const int tid   = threadIdx.x;
  const int lane  = tid & 31;
  const int wave  = tid >> 5;
  const int mw    = wave >> 4;      // 0..1
  const int nw    = wave & 15;      // 0..15
  const int l15   = lane & 15;
  const int khalf = lane >> 4;
  const int m0    = blockIdx.x * 32;
  const int nb    = nw * 64;

  v8f acc[4];
#pragma unroll
  for (int f = 0; f < 4; ++f)
#pragma unroll
    for (int r = 0; r < 8; ++r) acc[f][r] = 0.f;

  const __bf16* wbase  = WoutB + (size_t)(nb + l15) * EXPANDED + khalf * 16;
  const __bf16* aplane = s_g + (mw * 16 + l15) * 32;
  const size_t  fstride = (size_t)16 * EXPANDED;

#pragma unroll 1
  for (int k0 = 0; k0 < EXPANDED; k0 += 32) {
    __syncthreads();   // previous iteration's fragments fully consumed
    // --- stage gated = v * (w0*x[t-1] + w1*x[t] + w2*x[t+1] + cb) as bf16
    {
      const int e   = tid;                  // 1024 elements, 1 per thread
      const int r   = e >> 5;
      const int c   = e & 31;
      const int row = m0 + r;               // global bt index
      const int ch  = k0 + c;
      const int t   = row & (SEQ_T - 1);
      const size_t base = (size_t)row * TWO_E + ch;
      float xm = (t > 0)         ? (float)xv[base - TWO_E] : 0.f;
      float xc = (float)xv[base];
      float xp = (t < SEQ_T - 1) ? (float)xv[base + TWO_E] : 0.f;
      float vv = (float)xv[base + EXPANDED];
      const float* cw = conv_w + ch * 3;
      float y = fmaf(cw[0], xm, fmaf(cw[1], xc, fmaf(cw[2], xp, conv_b[ch])));
      s_g[r * 32 + c] = (__bf16)(vv * y);
    }
    __syncthreads();

    // --- A fragment from LDS (row-major 32x32 bf16 tile)
    v16bf a;
    {
      v8bf lo = *(const v8bf*)(aplane + khalf * 16);
      v8bf hi = *(const v8bf*)(aplane + khalf * 16 + 8);
#pragma unroll
      for (int i = 0; i < 8; ++i) { a[i] = lo[i]; a[8 + i] = hi[i]; }
    }
    const __bf16* wb = wbase + k0;
    __builtin_prefetch(wb + 32, 0, 1);
    // all 4 B fragments in flight before the WMMAs
    v16bf b0 = load_b_frag(wb);
    v16bf b1 = load_b_frag(wb + fstride);
    v16bf b2 = load_b_frag(wb + 2 * fstride);
    v16bf b3 = load_b_frag(wb + 3 * fstride);
    acc[0] = __builtin_amdgcn_wmma_f32_16x16x32_bf16(
        false, a, false, b0, (short)0, acc[0], false, false);
    acc[1] = __builtin_amdgcn_wmma_f32_16x16x32_bf16(
        false, a, false, b1, (short)0, acc[1], false, false);
    acc[2] = __builtin_amdgcn_wmma_f32_16x16x32_bf16(
        false, a, false, b2, (short)0, acc[2], false, false);
    acc[3] = __builtin_amdgcn_wmma_f32_16x16x32_bf16(
        false, a, false, b3, (short)0, acc[3], false, false);
  }

  // ---- epilogue: bias + RMSNorm over D_MODEL, f32 store
  if (tid < 32) s_ss[tid] = 0.f;
  __syncthreads();

  float ss[8];
#pragma unroll
  for (int r = 0; r < 8; ++r) ss[r] = 0.f;
#pragma unroll
  for (int f = 0; f < 4; ++f) {
    const float bias = b_out[nb + f * 16 + l15];
#pragma unroll
    for (int r = 0; r < 8; ++r) {
      float vv = acc[f][r] + bias;
      acc[f][r] = vv;
      ss[r] += vv * vv;
    }
  }
#pragma unroll
  for (int off = 1; off < 16; off <<= 1)
#pragma unroll
    for (int r = 0; r < 8; ++r) ss[r] += __shfl_xor(ss[r], off, 32);

  const int rbase = mw * 16 + khalf * 8;
  if (l15 == 0) {
#pragma unroll
    for (int r = 0; r < 8; ++r) atomicAdd(&s_ss[rbase + r], ss[r]);
  }
  __syncthreads();
  if (tid < 32) s_ss[tid] = rsqrtf(s_ss[tid] * (1.f / D_MODEL) + EPSILON);
  __syncthreads();

  float scale[8];
#pragma unroll
  for (int r = 0; r < 8; ++r) scale[r] = s_ss[rbase + r];

#pragma unroll
  for (int f = 0; f < 4; ++f) {
    const int col = nb + f * 16 + l15;
    const float g = nrm_w[col];
#pragma unroll
    for (int r = 0; r < 8; ++r) {
      out[(size_t)(m0 + rbase + r) * D_MODEL + col] = acc[f][r] * scale[r] * g;
    }
  }
}

// ============================================================================
extern "C" void kernel_launch(void* const* d_in, const int* in_sizes, int n_in,
                              void* d_out, int out_size, void* d_ws, size_t ws_size,
                              hipStream_t stream) {
  const float* u         = (const float*)d_in[0];
  const float* W_in      = (const float*)d_in[1];
  const float* b_in      = (const float*)d_in[2];
  const float* in_norm_w = (const float*)d_in[3];
  const float* conv_w    = (const float*)d_in[4];
  const float* conv_b    = (const float*)d_in[5];
  const float* W_out     = (const float*)d_in[6];
  const float* b_out     = (const float*)d_in[7];
  const float* out_norm_w= (const float*)d_in[8];
  float* out = (float*)d_out;

  // workspace layout (bf16): W_in (4MB) | W_out (2MB) | xv (64MB) ~= 70MB
  __bf16* WinB  = (__bf16*)d_ws;
  __bf16* WoutB = WinB  + (size_t)TWO_E * D_MODEL;
  __bf16* xvB   = WoutB + (size_t)D_MODEL * EXPANDED;

  const int nCvt = TWO_E * D_MODEL + D_MODEL * EXPANDED;
  k_cvt_weights<<<(nCvt + 255) / 256, 256, 0, stream>>>(W_in, W_out, WinB, WoutB);
  k_inproj <<<BT_TOTAL / 32, 1024, 0, stream>>>(u, WinB, b_in, in_norm_w, xvB);
  k_outproj<<<BT_TOTAL / 32, 1024, 0, stream>>>(xvB, WoutB, conv_w, conv_b,
                                                b_out, out_norm_w, out);
}
